// SKVMN_50173807952148
// MI455X (gfx1250) — matrix operand order repaired and checked by reading, not compiled, not run on hardware
//
#include <hip/hip_runtime.h>
#include <hip/hip_bf16.h>
#include <stdint.h>

// ---------------- model dims ----------------
#define BB   128
#define TT   48
#define DD   512
#define MMSL 64
#define NUMC 2000

typedef unsigned short     u16;
typedef unsigned long long u64;
typedef __attribute__((ext_vector_type(16))) __bf16 v16bf;
typedef __attribute__((ext_vector_type(8)))  float  v8f;
typedef int v4i __attribute__((vector_size(16)));   // matches async-LDS builtin param

union BFrag { v16bf v; uint4 q[2]; };
static_assert(sizeof(BFrag) == 32, "frag size");

// gfx1250 async global->LDS path (ASYNCcnt), with portable fallback
#if defined(__has_builtin)
#  if __has_builtin(__builtin_amdgcn_global_load_async_to_lds_b128)
#    define HAVE_ASYNC_LDS 1
#  endif
#endif
#ifndef HAVE_ASYNC_LDS
#  define HAVE_ASYNC_LDS 0
#endif

__device__ __forceinline__ void wait_async_le(int n) {
#if HAVE_ASYNC_LDS
#  if defined(__has_builtin) && __has_builtin(__builtin_amdgcn_s_wait_asynccnt)
  if (n == 0) __builtin_amdgcn_s_wait_asynccnt(0);
  else        __builtin_amdgcn_s_wait_asynccnt(1);
#  else
  if (n == 0) asm volatile("s_wait_asynccnt 0x0" ::: "memory");
  else        asm volatile("s_wait_asynccnt 0x1" ::: "memory");
#  endif
#else
  (void)n;  // fallback staging is synchronous; __syncthreads() orders LDS
#endif
}

__device__ __forceinline__ u16 f2bf(float f) {
  union { float f; unsigned u; } x; x.f = f;
  unsigned u = x.u;
  unsigned r = u + 0x7FFFu + ((u >> 16) & 1u);
  if ((u & 0x7FFFFFFFu) > 0x7F800000u) r = u | 0x00400000u;  // quiet NaN
  return (u16)(r >> 16);
}
__device__ __forceinline__ float sigf(float x) { return 1.f / (1.f + __expf(-x)); }

// ---------------- generic bf16 WMMA GEMM ----------------
// C[m,n] = act( sum_k A[m,k]*W[n,k] + bias[n] + extra[m,n] )
// A: (Mr,K) bf16 row-major (lda);  W: (Nc,K) bf16 row-major (ldw) == B^T
// Block = 8 waves; all waves share one 64-column panel -> weight tile staged
// in LDS once per block (async DMA, double buffered), A loaded direct.
// One wave -> 16x64 tile (4 x v_wmma_f32_16x16x32_bf16 per 32-wide K chunk).
#define SB_STRIDE 40  // u16 units per staged row (32 data + 8 pad; 80B, 16B aligned)

__global__ __launch_bounds__(256) void wmma_gemm_kernel(
    const u16* __restrict__ A, int lda,
    const u16* __restrict__ W, int ldw,
    float* __restrict__ Cf, u16* __restrict__ Cb, int ldc,
    const float* __restrict__ bias,
    const float* __restrict__ extra, int lde,
    int Mr, int Nc, int K, int act) {
  __shared__ u16 sB[2][64 * SB_STRIDE];
  (void)Nc;
  const int wave = threadIdx.x >> 5;
  const int lane = threadIdx.x & 31;
  int row0 = (blockIdx.y * 8 + wave) * 16;
  if (row0 > Mr - 16) row0 = Mr - 16;  // launches are exact; clamp keeps barriers safe
  const int col0 = blockIdx.x * 64;
  const int lm = lane & 15;
  const int hi = lane >> 4;

  // weight-panel staging: 256 threads x 16B == 64 rows x 32 k-values (bf16)
  const int srow = threadIdx.x >> 2;
  const int sseg = (threadIdx.x & 3) * 8;
  const u16* gstage = W + (size_t)(col0 + srow) * ldw + sseg;
  auto stage = [&](int buf, int k0) {
#if HAVE_ASYNC_LDS
    __builtin_amdgcn_global_load_async_to_lds_b128(
        (v4i*)(gstage + k0),
        (v4i*)(&sB[buf][srow * SB_STRIDE + sseg]),
        0, 0);
#else
    *(uint4*)(&sB[buf][srow * SB_STRIDE + sseg]) = *(const uint4*)(gstage + k0);
#endif
  };

  v8f acc[4] = {};
  // A fragment: 16-bit 16x32 layout — lanes 0-15: K {0..7,16..23}, lanes 16-31: {8..15,24..31}
  const u16* aRow = A + (size_t)(row0 + lm) * lda + hi * 8;

  const int nChunk = K >> 5;
  stage(0, 0);
  for (int ic = 0; ic < nChunk; ++ic) {
    const int k0 = ic << 5;
    const int cur = ic & 1;
    const bool pre = (ic + 1 < nChunk);
    if (pre) stage(cur ^ 1, k0 + 32);  // overlap next DMA with this chunk's WMMAs
    BFrag fa;
    fa.q[0] = *(const uint4*)(aRow + k0);
    fa.q[1] = *(const uint4*)(aRow + k0 + 16);
    wait_async_le(pre ? 1 : 0);  // current buffer's DMA done (prefetch may remain)
    __syncthreads();
    // B fragments from LDS: lane n=lm; lanes 0-15: K {0..7,16..23}, lanes 16-31: {8..15,24..31}
    BFrag fb[4];
#pragma unroll
    for (int j = 0; j < 4; ++j) {
      const u16* pb = &sB[cur][(j * 16 + lm) * SB_STRIDE + hi * 8];
      fb[j].q[0] = *(const uint4*)(pb);
      fb[j].q[1] = *(const uint4*)(pb + 16);
    }
#pragma unroll
    for (int j = 0; j < 4; ++j)
      acc[j] = __builtin_amdgcn_wmma_f32_16x16x32_bf16(
          false, fa.v, false, fb[j].v, (short)0, acc[j], false, false);
    __syncthreads();  // all waves done reading before this buffer is re-staged
  }

  // C/D layout: lane<16 -> N=lm, M=row0+v ; lane>=16 -> N=lm, M=row0+8+v
#pragma unroll
  for (int j = 0; j < 4; ++j) {
    const int n = col0 + j * 16 + lm;
#pragma unroll
    for (int v = 0; v < 8; ++v) {
      const int m = row0 + hi * 8 + v;
      float x = acc[j][v];
      if (bias)  x += bias[n];
      if (extra) x += extra[(size_t)m * lde + n];
      if (act == 1) x = tanhf(x);
      else if (act == 2) x = sigf(x);
      if (Cf) Cf[(size_t)m * ldc + n] = x;
      if (Cb) Cb[(size_t)m * ldc + n] = f2bf(x);
    }
  }
}

// ---------------- small utility kernels ----------------
__global__ void cast_bf_kernel(u16* dst, const float* src, long n) {
  long i = (long)blockIdx.x * blockDim.x + threadIdx.x;
  if (i < n) dst[i] = f2bf(src[i]);
}

// dst (rows, 2d) bf16 = [s0 | s1] per row  (for Wcat = [W_ih | W_hh])
__global__ void cast_concat_kernel(u16* dst, const float* s0, const float* s1,
                                   int rows, int d) {
  long i = (long)blockIdx.x * blockDim.x + threadIdx.x;
  long total = (long)rows * 2 * d;
  if (i >= total) return;
  int rr = (int)(i / (2 * d));
  int cc = (int)(i % (2 * d));
  float v = (cc < d) ? s0[(size_t)rr * d + cc] : s1[(size_t)rr * d + (cc - d)];
  dst[i] = f2bf(v);
}

__global__ void vec_add_kernel(float* dst, const float* a, const float* b, int n) {
  int i = blockIdx.x * blockDim.x + threadIdx.x;
  if (i < n) dst[i] = a[i] + b[i];
}

// dst[r, :] = src[r % srcrows, :]
__global__ void tile_rows_kernel(float* dst, const float* src, int rows,
                                 int srcrows, int d) {
  long i = (long)blockIdx.x * blockDim.x + threadIdx.x;
  long total = (long)rows * d;
  if (i >= total) return;
  int rr = (int)(i / d);
  int cc = (int)(i % d);
  dst[i] = src[(size_t)(rr % srcrows) * d + cc];
}

// gather k_table[q] and x_table[q + NUMC*r] -> bf16 rows (row = b*T+t)
__global__ __launch_bounds__(256) void embed_kernel(
    const int* __restrict__ q, const int* __restrict__ r,
    const float* __restrict__ ktab, const float* __restrict__ xtab,
    u16* __restrict__ kbf, u16* __restrict__ ybf) {
  int row = blockIdx.x;
  const float* kp = ktab + (size_t)q[row] * DD;
  const float* yp = xtab + (size_t)(q[row] + NUMC * r[row]) * DD;
  for (int d = threadIdx.x; d < DD; d += 256) {
    kbf[(size_t)row * DD + d] = f2bf(kp[d]);
    ybf[(size_t)row * DD + d] = f2bf(yp[d]);
  }
}

// fp32 addressing scores s[row,m] = k_table[q[row]] . Mk[m]  (discrete path stays fp32)
__global__ __launch_bounds__(256) void addr_scores_kernel(
    const int* __restrict__ q, const float* __restrict__ ktab,
    const float* __restrict__ Mk, float* __restrict__ s) {
  int row = blockIdx.x;
  __shared__ float kr[DD];
  const float* kp = ktab + (size_t)q[row] * DD;
  for (int d = threadIdx.x; d < DD; d += 256) kr[d] = kp[d];
  __syncthreads();
  int wid = threadIdx.x >> 5, lane = threadIdx.x & 31;
  for (int m = wid; m < MMSL; m += 8) {
    const float* mk = Mk + (size_t)m * DD;
    float acc = 0.f;
    for (int d = lane; d < DD; d += 32) acc += kr[d] * mk[d];
    for (int o = 16; o > 0; o >>= 1) acc += __shfl_xor(acc, o, 32);
    if (lane == 0) s[(size_t)row * MMSL + m] = acc;
  }
}

// softmax over M + triangular membership -> 2-bit iv codes packed in 128-bit sig
__global__ void softmax_sig_kernel(const float* __restrict__ s, float* __restrict__ w,
                                   u64* __restrict__ sigA, u64* __restrict__ sigB) {
  int row = blockIdx.x * blockDim.x + threadIdx.x;
  if (row >= BB * TT) return;
  const float* sr = s + (size_t)row * MMSL;
  float mx = sr[0];
  for (int m = 1; m < MMSL; ++m) mx = fmaxf(mx, sr[m]);
  float sum = 0.f;
  for (int m = 0; m < MMSL; ++m) sum += __expf(sr[m] - mx);
  float inv = 1.f / sum;
  u64 lo = 0, hic = 0;
  for (int m = 0; m < MMSL; ++m) {
    float wv = __expf(sr[m] - mx) * inv;
    w[(size_t)row * MMSL + m] = wv;
    float t1 = (wv - 0.075f) / (0.088f - 0.075f);
    float t2 = (1.0f - wv) / (1.0f - 0.088f);
    float mem = fmaxf(fminf(t1, t2), 0.f);
    u64 code = (mem >= 0.6f) ? 2ull : ((mem >= 0.1f) ? 1ull : 0ull);
    if (m < 32) lo |= code << (2 * m); else hic |= code << (2 * (m - 32));
  }
  sigA[row] = lo; sigB[row] = hic;
}

// most recent j<i with identical identity vector, else i-1
__global__ void src_scan_kernel(const u64* __restrict__ sigA,
                                const u64* __restrict__ sigB, int* __restrict__ src) {
  int idx = blockIdx.x * blockDim.x + threadIdx.x;
  if (idx >= BB * TT) return;
  int b = idx / TT, i = idx % TT;
  u64 la = sigA[idx], lb = sigB[idx];
  const u64* bA = sigA + (size_t)b * TT;
  const u64* bB = sigB + (size_t)b * TT;
  int sv = i - 1;
  for (int j = i - 1; j >= 0; --j)
    if (bA[j] == la && bB[j] == lb) { sv = j; break; }
  src[idx] = sv;
}

// read_t[b,:] = sum_m w[b,t,m] * mem[b,m,:]  -> bf16
__global__ __launch_bounds__(256) void read_kernel(
    const float* __restrict__ w, const float* __restrict__ mem,
    u16* __restrict__ readbf, int t) {
  int b = blockIdx.x;
  __shared__ float ws_[MMSL];
  if (threadIdx.x < MMSL) ws_[threadIdx.x] = w[((size_t)b * TT + t) * MMSL + threadIdx.x];
  __syncthreads();
  for (int d = threadIdx.x; d < DD; d += 256) {
    const float* mb = mem + (size_t)b * MMSL * DD + d;
    float acc = 0.f;
#pragma unroll 8
    for (int m = 0; m < MMSL; ++m) acc += ws_[m] * mb[(size_t)m * DD];
    readbf[(size_t)b * DD + d] = f2bf(acc);
  }
}

// mem = mem * (1 - w (x) sigmoid(ea[:,0:512])) + w (x) tanh(ea[:,512:1024])
__global__ __launch_bounds__(256) void update_kernel(
    const float* __restrict__ w, const float* __restrict__ ea,
    float* __restrict__ mem, int t) {
  int b = blockIdx.x;
  __shared__ float ws_[MMSL];
  if (threadIdx.x < MMSL) ws_[threadIdx.x] = w[((size_t)b * TT + t) * MMSL + threadIdx.x];
  __syncthreads();
  for (int d = threadIdx.x; d < DD; d += 256) {
    float e = sigf(ea[(size_t)b * 2 * DD + d]);
    float a = tanhf(ea[(size_t)b * 2 * DD + DD + d]);
    float* mb = mem + (size_t)b * MMSL * DD + d;
#pragma unroll 8
    for (int m = 0; m < MMSL; ++m) {
      float wm = ws_[m];
      float v = mb[(size_t)m * DD];
      mb[(size_t)m * DD] = v * (1.f - wm * e) + wm * a;
    }
  }
}

// A = [ f_t (bf16) | bf16(H[src+1]) ]  (128 x 1024)
__global__ __launch_bounds__(256) void lstm_pack_kernel(
    const u16* __restrict__ ftbf, const float* __restrict__ H,
    const int* __restrict__ src, u16* __restrict__ Abuf, int t) {
  int b = blockIdx.x;
  int sv = src[(size_t)b * TT + t];
  const float* hrow = H + ((size_t)(sv + 1) * BB + b) * DD;
  const u16* frow = ftbf + ((size_t)t * BB + b) * DD;
  for (int d = threadIdx.x; d < DD; d += 256) {
    Abuf[(size_t)b * 2 * DD + d] = frow[d];
    Abuf[(size_t)b * 2 * DD + DD + d] = f2bf(hrow[d]);
  }
}

// LSTM pointwise: gates (128,2048) [i|f|g|o], c_in = C[src+1]; writes H/C slot t+1
__global__ __launch_bounds__(256) void lstm_point_kernel(
    const float* __restrict__ gates, const int* __restrict__ src,
    float* __restrict__ H, float* __restrict__ C, int t) {
  int b = blockIdx.x;
  int sv = src[(size_t)b * TT + t];
  for (int d = threadIdx.x; d < DD; d += 256) {
    float gi = gates[(size_t)b * 4 * DD + d];
    float gf = gates[(size_t)b * 4 * DD + DD + d];
    float gg = gates[(size_t)b * 4 * DD + 2 * DD + d];
    float go = gates[(size_t)b * 4 * DD + 3 * DD + d];
    float cin = C[((size_t)(sv + 1) * BB + b) * DD + d];
    float c = sigf(gf) * cin + sigf(gi) * tanhf(gg);
    float h = sigf(go) * tanhf(c);
    H[((size_t)(t + 1) * BB + b) * DD + d] = h;
    C[((size_t)(t + 1) * BB + b) * DD + d] = c;
  }
}

// p[b,t] = sigmoid(H[t+1,b,:] . W_p + b_p)  — one wave per output
__global__ __launch_bounds__(256) void head_kernel(
    const float* __restrict__ H, const float* __restrict__ Wp,
    const float* __restrict__ bp, float* __restrict__ out) {
  int wid = threadIdx.x >> 5, lane = threadIdx.x & 31;
  int idx = blockIdx.x * 8 + wid;
  if (idx >= BB * TT) return;
  int b = idx / TT, t = idx % TT;
  const float* h = H + ((size_t)(t + 1) * BB + b) * DD;
  float acc = 0.f;
  for (int d = lane; d < DD; d += 32) acc += h[d] * Wp[d];
  for (int o = 16; o > 0; o >>= 1) acc += __shfl_xor(acc, o, 32);
  if (lane == 0) out[idx] = sigf(acc + bp[0]);
}

// ---------------- host driver ----------------
extern "C" void kernel_launch(void* const* d_in, const int* in_sizes, int n_in,
                              void* d_out, int out_size, void* d_ws, size_t ws_size,
                              hipStream_t stream) {
  (void)in_sizes; (void)n_in; (void)out_size; (void)ws_size;
  const int*   q     = (const int*)d_in[0];
  const int*   r     = (const int*)d_in[1];
  const float* Mk    = (const float*)d_in[2];
  const float* Mv0   = (const float*)d_in[3];
  const float* ktab  = (const float*)d_in[4];
  const float* xtab  = (const float*)d_in[5];
  const float* W_e   = (const float*)d_in[6];   // b_e=d_in[7] (zeros)
  const float* W_add = (const float*)d_in[8];   // b_add=d_in[9] (zeros)
  const float* W_a   = (const float*)d_in[10];
  const float* b_a   = (const float*)d_in[11];
  const float* W_f   = (const float*)d_in[12];
  const float* b_f   = (const float*)d_in[13];
  const float* W_ih  = (const float*)d_in[14];
  const float* W_hh  = (const float*)d_in[15];
  const float* b_ih  = (const float*)d_in[16];
  const float* b_hh  = (const float*)d_in[17];
  const float* hx0   = (const float*)d_in[18];
  const float* cx0   = (const float*)d_in[19];
  const float* W_p   = (const float*)d_in[20];
  const float* b_p   = (const float*)d_in[21];
  float* out = (float*)d_out;

  // workspace bump allocator (256B aligned)
  char* base = (char*)d_ws;
  size_t off = 0;
  auto alloc = [&](size_t bytes) -> void* {
    off = (off + 255) & ~(size_t)255;
    void* p = base + off; off += bytes; return p;
  };
  const int ROWS = BB * TT;  // 6144
  u16*  wfbf   = (u16*)alloc((size_t)DD * 2 * DD * 2);        // (512,1024) bf16
  u16*  wabf   = (u16*)alloc((size_t)DD * 2 * DD * 2);        // (512,1024) bf16
  u16*  weabf  = (u16*)alloc((size_t)2 * DD * DD * 2);        // [W_e;W_add] (1024,512) bf16
  u16*  wcatbf = (u16*)alloc((size_t)4 * DD * 2 * DD * 2);    // [W_ih|W_hh] (2048,1024) bf16
  float* bihhh = (float*)alloc(4 * DD * 4);                   // b_ih + b_hh
  u16*  kbf    = (u16*)alloc((size_t)ROWS * DD * 2);
  u16*  ybf    = (u16*)alloc((size_t)ROWS * DD * 2);
  float* s     = (float*)alloc((size_t)ROWS * MMSL * 4);
  float* w     = (float*)alloc((size_t)ROWS * MMSL * 4);
  u64*  sigA   = (u64*)alloc((size_t)ROWS * 8);
  u64*  sigB   = (u64*)alloc((size_t)ROWS * 8);
  int*  src    = (int*)alloc((size_t)ROWS * 4);
  float* kf    = (float*)alloc((size_t)ROWS * DD * 4);        // k@Wf2^T + b_f
  float* ya    = (float*)alloc((size_t)ROWS * DD * 4);        // y@Wa2^T + b_a
  float* mem   = (float*)alloc((size_t)BB * MMSL * DD * 4);
  u16*  readbf = (u16*)alloc((size_t)BB * DD * 2);
  u16*  ftbf   = (u16*)alloc((size_t)TT * BB * DD * 2);
  u16*  webf   = (u16*)alloc((size_t)BB * DD * 2);
  float* ea    = (float*)alloc((size_t)BB * 2 * DD * 4);
  float* H     = (float*)alloc((size_t)(TT + 1) * BB * DD * 4);
  float* C     = (float*)alloc((size_t)(TT + 1) * BB * DD * 4);
  u16*  Abuf   = (u16*)alloc((size_t)BB * 2 * DD * 2);
  float* gates = (float*)alloc((size_t)BB * 4 * DD * 4);

  auto blocks = [](long n, int bs) { return (unsigned)((n + bs - 1) / bs); };
  auto gemm = [&](const u16* A, int lda, const u16* W, int ldw,
                  float* Cf, u16* Cb, int ldc,
                  const float* bias, const float* extra, int lde,
                  int Mr, int Nc, int K, int act) {
    dim3 g((unsigned)(Nc / 64), (unsigned)((Mr / 16 + 7) / 8));
    wmma_gemm_kernel<<<g, 256, 0, stream>>>(A, lda, W, ldw, Cf, Cb, ldc,
                                            bias, extra, lde, Mr, Nc, K, act);
  };

  // ---- weight conversions (bf16) ----
  cast_bf_kernel<<<blocks((long)DD * 2 * DD, 256), 256, 0, stream>>>(wfbf, W_f, (long)DD * 2 * DD);
  cast_bf_kernel<<<blocks((long)DD * 2 * DD, 256), 256, 0, stream>>>(wabf, W_a, (long)DD * 2 * DD);
  cast_bf_kernel<<<blocks((long)DD * DD, 256), 256, 0, stream>>>(weabf, W_e, (long)DD * DD);
  cast_bf_kernel<<<blocks((long)DD * DD, 256), 256, 0, stream>>>(weabf + (size_t)DD * DD, W_add, (long)DD * DD);
  cast_concat_kernel<<<blocks((long)4 * DD * 2 * DD, 256), 256, 0, stream>>>(wcatbf, W_ih, W_hh, 4 * DD, DD);
  vec_add_kernel<<<blocks(4 * DD, 256), 256, 0, stream>>>(bihhh, b_ih, b_hh, 4 * DD);

  // ---- embeddings, addressing (fp32 discrete path), hop sources ----
  embed_kernel<<<ROWS, 256, 0, stream>>>(q, r, ktab, xtab, kbf, ybf);
  addr_scores_kernel<<<ROWS, 256, 0, stream>>>(q, ktab, Mk, s);
  softmax_sig_kernel<<<blocks(ROWS, 256), 256, 0, stream>>>(s, w, sigA, sigB);
  src_scan_kernel<<<blocks(ROWS, 256), 256, 0, stream>>>(sigA, sigB, src);

  // ---- recurrence-independent big GEMMs (WMMA) ----
  // kf = k @ W_f[:,512:]^T + b_f ;  ya = y @ W_a[:,512:]^T + b_a
  gemm(kbf, DD, wfbf + DD, 2 * DD, kf, nullptr, DD, b_f, nullptr, 0, ROWS, DD, DD, 0);
  gemm(ybf, DD, wabf + DD, 2 * DD, ya, nullptr, DD, b_a, nullptr, 0, ROWS, DD, DD, 0);

  // ---- state init ----
  tile_rows_kernel<<<blocks((long)BB * DD, 256), 256, 0, stream>>>(H, hx0, BB, 1, DD);
  tile_rows_kernel<<<blocks((long)BB * DD, 256), 256, 0, stream>>>(C, cx0, BB, 1, DD);
  tile_rows_kernel<<<blocks((long)BB * MMSL * DD, 256), 256, 0, stream>>>(mem, Mv0, BB * MMSL, MMSL, DD);

  // ---- DKVMN value-memory recurrence ----
  for (int t = 0; t < TT; ++t) {
    read_kernel<<<BB, 256, 0, stream>>>(w, mem, readbf, t);
    // f = tanh(read@Wf1^T + kf_t)   (bf16 out -> ftbf[t])
    gemm(readbf, DD, wfbf, 2 * DD, nullptr, ftbf + (size_t)t * BB * DD, DD,
         nullptr, kf + (size_t)t * DD, TT * DD, BB, DD, DD, 1);
    // we = f@Wa1^T + ya_t           (bf16 out)
    gemm(ftbf + (size_t)t * BB * DD, DD, wabf, 2 * DD, nullptr, webf, DD,
         nullptr, ya + (size_t)t * DD, TT * DD, BB, DD, DD, 0);
    // ea = we @ [W_e;W_add]^T       (pre-activation, fp32)
    gemm(webf, DD, weabf, DD, ea, nullptr, 2 * DD,
         nullptr, nullptr, 0, BB, 2 * DD, DD, 0);
    update_kernel<<<BB, 256, 0, stream>>>(w, ea, mem, t);
  }

  // ---- hop-LSTM ----
  for (int t = 0; t < TT; ++t) {
    lstm_pack_kernel<<<BB, 256, 0, stream>>>(ftbf, H, src, Abuf, t);
    // gates = [f|h_src] @ [W_ih|W_hh]^T + (b_ih+b_hh)
    gemm(Abuf, 2 * DD, wcatbf, 2 * DD, gates, nullptr, 4 * DD,
         bihhh, nullptr, 0, BB, 4 * DD, 2 * DD, 0);
    lstm_point_kernel<<<BB, 256, 0, stream>>>(gates, src, H, C, t);
  }

  // ---- prediction head ----
  head_kernel<<<blocks(ROWS, 8), 256, 0, stream>>>(H, W_p, b_p, out);
}
